// MTHiSAN_36412732736171
// MI455X (gfx1250) — matrix operand order repaired and checked by reading, not compiled, not run on hardware
//
#include <hip/hip_runtime.h>
#include <math.h>

// ---------------- problem constants ----------------
constexpr int kEMB   = 300;
constexpr int kAD    = 400;     // HEADS * DPH
constexpr int kHEADS = 8;
constexpr int kDPH   = 50;
constexpr int kWPL   = 32;      // words per line
constexpr int kLINES = 128;     // lines per doc
constexpr int kBATCH = 16;
constexpr int kNLINE = kBATCH * kLINES;   // 2048
constexpr int kVOCAB = 30000;
constexpr float kNEG  = 1e7f;
constexpr float kINVS = 0.14142135623730951f; // 1/sqrt(50)

// ---------------- WMMA plumbing (gfx1250) ----------------
typedef __attribute__((ext_vector_type(16))) __bf16 v16bf;
typedef __attribute__((ext_vector_type(8)))  float  v8f;
typedef __attribute__((ext_vector_type(4)))  unsigned int v4u;
typedef __attribute__((ext_vector_type(8)))  int    v8i_;
typedef __attribute__((ext_vector_type(4)))  int    v4i_;

__device__ __forceinline__ v8f wmma_bf16(v16bf a, v16bf b, v8f c) {
  return __builtin_amdgcn_wmma_f32_16x16x32_bf16(
      false, a, false, b, (short)0, c, false, false);
}

// A-matrix 16x32 bf16 fragment per ISA 7.12.2
template <class F>
__device__ __forceinline__ v16bf make_afrag(F&& f, int lane) {
  v16bf a;
  const int h = (lane >> 4) & 1, m = lane & 15;
  #pragma unroll
  for (int s = 0; s < 8; ++s) {
    const int kb = (s < 4 ? 2 * s : 16 + 2 * (s - 4)) + 8 * h;
    a[2 * s]     = f(m, kb);
    a[2 * s + 1] = f(m, kb + 1);
  }
  return a;
}

// B-matrix 32x16 bf16 fragment: lanes = N column, same K order as A fragment
template <class F>
__device__ __forceinline__ v16bf make_bfrag(F&& f, int lane) {
  v16bf b;
  const int h = (lane >> 4) & 1, n = lane & 15;
  #pragma unroll
  for (int s = 0; s < 8; ++s) {
    const int k = (s < 4 ? 2 * s : 16 + 2 * (s - 4)) + 8 * h;
    b[2 * s]     = f(k, n);
    b[2 * s + 1] = f(k + 1, n);
  }
  return b;
}

// C/D 16x16 f32: lane half h, col n = lane&15, VGPR r -> row r + 8h
template <class F>
__device__ __forceinline__ void store_cfrag(v8f c, int lane, F&& st) {
  const int h = (lane >> 4) & 1, n = lane & 15;
  #pragma unroll
  for (int r = 0; r < 8; ++r) st(r + 8 * h, n, c[r]);
}

__device__ __forceinline__ float elu(float x) {
  return x > 0.f ? x : (__expf(x) - 1.f);
}

// ---------------- Tensor Data Mover: gather-mode embedding fetch ----------------
#if __has_builtin(__builtin_amdgcn_tensor_load_to_lds)
#define HAVE_TDM 1
#else
#define HAVE_TDM 0
#endif

#if HAVE_TDM
// Gather 16 rows of `rowlen` f32 elements from a [nrows x rowlen] tensor into
// LDS at byte offset lds_off.  packed[8] = sixteen 16-bit row indices.
__device__ __forceinline__ void tdm_gather16_f32(
    unsigned lds_off, const float* gbase, const unsigned* packed,
    int rowlen, int nrows) {
  const unsigned long long ga = (unsigned long long)(uintptr_t)gbase;
  v4u g0;
  g0[0] = 1u | (1u << 31);                         // count=1, gather_mode, 16-bit idx
  g0[1] = lds_off;                                 // LDS destination (bytes)
  g0[2] = (unsigned)ga;                            // global_addr[31:0]
  g0[3] = (unsigned)((ga >> 32) & 0x01FFFFFFu) | (2u << 30);  // addr[56:32] | type=2
  v8i_ g1;
  g1[0] = (2 << 16);                               // data_size = 4 bytes
  g1[1] = (rowlen & 0xFFFF) << 16;                 // tensor_dim0 [63:48]
  g1[2] = (nrows & 0xFFFF) << 16;                  // tensor_dim0 hi=0 | tensor_dim1 lo
  g1[3] = ((nrows >> 16) & 0xFFFF) | ((rowlen & 0xFFFF) << 16); // dim1 hi | tile_dim0
  g1[4] = 16;                                      // tile_dim1 = #valid indices
  g1[5] = rowlen;                                  // tensor_dim0_stride [191:160]
  g1[6] = 0;
  g1[7] = 0;
  v4i_ g2, g3;
  #pragma unroll
  for (int i = 0; i < 4; ++i) { g2[i] = (int)packed[i]; g3[i] = (int)packed[4 + i]; }
#if defined(__clang_major__) && __clang_major__ >= 23
  v8i_ g23;
  #pragma unroll
  for (int i = 0; i < 4; ++i) { g23[i] = g2[i]; g23[4 + i] = g3[i]; }
  __builtin_amdgcn_tensor_load_to_lds(g0, g1, g2, g3, g23, 0);
#else
  __builtin_amdgcn_tensor_load_to_lds(g0, g1, g2, g3, 0);
#endif
}
#endif

// =====================================================================
// Kernel 0: one-shot fp32 -> bf16 weight conversion (runs once per launch)
// =====================================================================
__global__ __launch_bounds__(256) void cvt_weights_kernel(
    const float* __restrict__ wqw, const float* __restrict__ wkw,
    const float* __restrict__ wvw, const float* __restrict__ lqw,
    const float* __restrict__ lkw, const float* __restrict__ lvw,
    __bf16* __restrict__ wqB, __bf16* __restrict__ wkB, __bf16* __restrict__ wvB,
    __bf16* __restrict__ lqB, __bf16* __restrict__ lkB, __bf16* __restrict__ lvB) {
  const int stride = gridDim.x * 256;
  for (int j = blockIdx.x * 256 + threadIdx.x; j < kEMB * kAD; j += stride) {
    wqB[j] = (__bf16)wqw[j]; wkB[j] = (__bf16)wkw[j]; wvB[j] = (__bf16)wvw[j];
  }
  for (int j = blockIdx.x * 256 + threadIdx.x; j < kAD * kAD; j += stride) {
    lqB[j] = (__bf16)lqw[j]; lkB[j] = (__bf16)lkw[j]; lvB[j] = (__bf16)lvw[j];
  }
}

// =====================================================================
// Kernel 1: per-line fused  TDM-gather -> QKV projection (WMMA) -> word
// attention (wave==head) -> target attention -> line embedding (bf16)
// grid = 2048 blocks x 256 threads, dynamic LDS = 131328 B
// LDS layout (bytes):
//   0      sQ   32x400 bf16   (phase 0 aliases this as f32 gather staging)
//   25600  sK   32x400 bf16
//   51200  sV   32x400 bf16
//   76800  sA   32x320 bf16   (phase 2 aliases as sW bf16 softmax weights)
//   97280  sS   8x32x32 f32   (later aliased as sWatt bf16)
//   130048 sMask 64 int
//   130304 sRed  8x32 f32
// =====================================================================
__global__ __launch_bounds__(256) void word_kernel(
    const int* __restrict__ docs, const float* __restrict__ emb,
    const __bf16* __restrict__ wqB, const float* __restrict__ wqb,
    const __bf16* __restrict__ wkB, const float* __restrict__ wkb,
    const __bf16* __restrict__ wvB, const float* __restrict__ wvb,
    const float* __restrict__ wtgt,
    __bf16* __restrict__ lineEb, int* __restrict__ maskL) {
  extern __shared__ char smemW[];
  __bf16* sQ = (__bf16*)smemW;
  __bf16* sK = sQ + kWPL * kAD;
  __bf16* sV = sK + kWPL * kAD;
  __bf16* sA = sV + kWPL * kAD;
  float*  sS = (float*)(sA + kWPL * 320);
  __bf16* sW = (__bf16*)sA;
  __bf16* sWatt = (__bf16*)sS;
  int*    sMask = (int*)(sS + kHEADS * kWPL * kWPL);
  float*  sRed  = (float*)(sMask + 64);

  const int line = blockIdx.x;
  const int tid  = threadIdx.x;
  const int lane = tid & 31;
  const int wave = tid >> 5;

  // ---- phase 0: tokens + word mask
  if (tid < kWPL) {
    const int tok   = docs[line * kWPL + tid];
    sMask[tid]      = (tok != 0) ? 1 : 0;
    sMask[32 + tid] = tok;
  }

#if HAVE_TDM
  // TDM gather: two descriptors x 16 rows of 300 f32 -> LDS staging @ offset 0
  if (wave == 0) {
    unsigned pk[16];
    #pragma unroll
    for (int i = 0; i < 16; ++i) {
      const unsigned lo = (unsigned)docs[line * kWPL + 2 * i] & 0xFFFFu;
      const unsigned hi = (unsigned)docs[line * kWPL + 2 * i + 1] & 0xFFFFu;
      pk[i] = lo | (hi << 16);
    }
    tdm_gather16_f32(0u,     emb, pk,     kEMB, kVOCAB);
    tdm_gather16_f32(19200u, emb, pk + 8, kEMB, kVOCAB);
    __builtin_amdgcn_s_wait_tensorcnt(0);
  }
  __syncthreads();
  {
    const float* stage = (const float*)smemW;   // 32 x 300 f32, rows compacted
    for (int i = tid; i < kWPL * 320; i += 256) {
      const int m = i / 320, k = i - m * 320;
      sA[i] = (__bf16)((k < kEMB) ? stage[m * kEMB + k] : 0.f);
    }
  }
#else
  __syncthreads();
  for (int i = tid; i < kWPL * 320; i += 256) {
    const int m = i / 320, k = i - m * 320;
    float v = 0.f;
    if (k < kEMB) v = emb[sMask[32 + m] * kEMB + k];
    sA[i] = (__bf16)v;
  }
#endif
  __syncthreads();

  // ---- phase 1: Q/K/V = ELU(we @ W + b); fixed pointers, A-frag reused x3
  for (int t = wave; t < 50; t += 8) {          // 2 Mtiles x 25 Ntiles
    const int mt = t / 25, nt = t - mt * 25;
    const int m0 = mt * 16, n0 = nt * 16;
    v8f aq = {}, ak = {}, av = {};
    for (int kc = 0; kc < 10; ++kc) {           // K = 300 padded to 320
      const int k0 = kc * 32;
      v16bf a = make_afrag([&](int m, int k) {
        return sA[(m0 + m) * 320 + k0 + k]; }, lane);
      v16bf bq = make_bfrag([&](int k, int n) {
        const int kk = k0 + k;
        return (kk < kEMB) ? wqB[kk * kAD + n0 + n] : (__bf16)0.f; }, lane);
      aq = wmma_bf16(a, bq, aq);
      v16bf bk = make_bfrag([&](int k, int n) {
        const int kk = k0 + k;
        return (kk < kEMB) ? wkB[kk * kAD + n0 + n] : (__bf16)0.f; }, lane);
      ak = wmma_bf16(a, bk, ak);
      v16bf bv = make_bfrag([&](int k, int n) {
        const int kk = k0 + k;
        return (kk < kEMB) ? wvB[kk * kAD + n0 + n] : (__bf16)0.f; }, lane);
      av = wmma_bf16(a, bv, av);
    }
    store_cfrag(aq, lane, [&](int m, int n, float v) {
      sQ[(m0 + m) * kAD + n0 + n] = (__bf16)elu(v + wqb[n0 + n]); });
    store_cfrag(ak, lane, [&](int m, int n, float v) {
      sK[(m0 + m) * kAD + n0 + n] = (__bf16)elu(v + wkb[n0 + n]); });
    store_cfrag(av, lane, [&](int m, int n, float v) {
      sV[(m0 + m) * kAD + n0 + n] = (__bf16)elu(v + wvb[n0 + n]); });
  }
  __syncthreads();

  // ---- phase 2: word attention, wave == head
  const int h = wave;
  for (int mt = 0; mt < 2; ++mt)
    for (int nt = 0; nt < 2; ++nt) {
      const int m0 = mt * 16, n0 = nt * 16;
      v8f acc = {};
      for (int kc = 0; kc < 2; ++kc) {          // D = 50 padded to 64
        const int k0 = kc * 32;
        v16bf a = make_afrag([&](int m, int k) {
          const int kk = k0 + k;
          return (kk < kDPH) ? sQ[(m0 + m) * kAD + h * kDPH + kk] : (__bf16)0.f; }, lane);
        v16bf b = make_bfrag([&](int k, int n) {
          const int kk = k0 + k;
          return (kk < kDPH) ? sK[(n0 + n) * kAD + h * kDPH + kk] : (__bf16)0.f; }, lane);
        acc = wmma_bf16(a, b, acc);
      }
      store_cfrag(acc, lane, [&](int m, int n, float v) {
        float s = v * kINVS;
        if (!sMask[n0 + n]) s -= kNEG;
        sS[(h * kWPL + m0 + m) * kWPL + n0 + n] = s;
      });
    }
  __syncthreads();
  {
    const int row = lane;
    float buf[kWPL], mx = -3.4e38f;
    #pragma unroll
    for (int j = 0; j < kWPL; ++j) { buf[j] = sS[(h * kWPL + row) * kWPL + j]; mx = fmaxf(mx, buf[j]); }
    float sum = 0.f;
    #pragma unroll
    for (int j = 0; j < kWPL; ++j) { buf[j] = __expf(buf[j] - mx); sum += buf[j]; }
    const float rs = (sMask[row] ? 1.f : 0.f) / sum;
    #pragma unroll
    for (int j = 0; j < kWPL; ++j) sW[(h * kWPL + row) * kWPL + j] = (__bf16)(buf[j] * rs);
  }
  __syncthreads();
  for (int mt = 0; mt < 2; ++mt)
    for (int nt = 0; nt < 4; ++nt) {            // N = 50 padded to 64
      const int m0 = mt * 16, n0 = nt * 16;
      v16bf a = make_afrag([&](int m, int k) {
        return sW[(h * kWPL + m0 + m) * kWPL + k]; }, lane);
      v16bf b = make_bfrag([&](int k, int n) {
        const int nn = n0 + n;
        return (nn < kDPH) ? sV[k * kAD + h * kDPH + nn] : (__bf16)0.f; }, lane);
      v8f acc = {};
      acc = wmma_bf16(a, b, acc);
      store_cfrag(acc, lane, [&](int m, int n, float v) {
        const int nn = n0 + n;
        if (nn < kDPH) {
          const float qm = sMask[m0 + m] ? 1.f : 0.f;
          sWatt[(h * kWPL + m0 + m) * kDPH + nn] = (__bf16)(v * qm);
        }
      });
    }
  __syncthreads();

  // ---- phase 3: target (1-query) attention, wave32 shuffle softmax
  {
    const int j = lane;
    float s = 0.f;
    for (int d = 0; d < kDPH; ++d)
      s += wtgt[h * kDPH + d] * (float)sK[j * kAD + h * kDPH + d];
    s *= kINVS;
    if (!sMask[j]) s -= kNEG;
    float mx = s;
    #pragma unroll
    for (int off = 16; off > 0; off >>= 1) mx = fmaxf(mx, __shfl_xor(mx, off, 32));
    const float e = __expf(s - mx);
    float sum = e;
    #pragma unroll
    for (int off = 16; off > 0; off >>= 1) sum += __shfl_xor(sum, off, 32);
    sRed[h * kWPL + j] = e / sum;
  }
  __syncthreads();
  {
    int lm = 0;
    for (int j = 0; j < kWPL; ++j) lm |= sMask[j];
    for (int d = lane; d < kDPH; d += 32) {
      float acc = 0.f;
      for (int j = 0; j < kWPL; ++j)
        acc += sRed[h * kWPL + j] * (float)sWatt[(h * kWPL + j) * kDPH + d];
      lineEb[line * kAD + h * kDPH + d] = (__bf16)(lm ? acc : 0.f);
    }
    if (tid == 0) maskL[line] = lm;
  }
}

// =====================================================================
// Kernel 2: line projections lq/lk/lv = ELU(lineE @ W + b); all-bf16 operands
// one wave per 16x16 tile, A-fragment shared across 3 weight matrices
// =====================================================================
__global__ __launch_bounds__(256) void lineproj_kernel(
    const __bf16* __restrict__ lineEb,
    const __bf16* __restrict__ lqB, const float* __restrict__ lqb,
    const __bf16* __restrict__ lkB, const float* __restrict__ lkb,
    const __bf16* __restrict__ lvB, const float* __restrict__ lvb,
    __bf16* __restrict__ lq, __bf16* __restrict__ lk, __bf16* __restrict__ lv) {
  const int gw = blockIdx.x * 8 + (threadIdx.x >> 5);
  const int lane = threadIdx.x & 31;
  if (gw >= 128 * 25) return;
  const int mt = gw / 25, nt = gw - mt * 25;
  const int m0 = mt * 16, n0 = nt * 16;
  v8f a0 = {}, a1 = {}, a2 = {};
  for (int kc = 0; kc < 13; ++kc) {             // K = 400 padded to 416
    const int k0 = kc * 32;
    v16bf a = make_afrag([&](int m, int k) {
      const int kk = k0 + k;
      return (kk < kAD) ? lineEb[(m0 + m) * kAD + kk] : (__bf16)0.f; }, lane);
    v16bf b0 = make_bfrag([&](int k, int n) {
      const int kk = k0 + k;
      return (kk < kAD) ? lqB[kk * kAD + n0 + n] : (__bf16)0.f; }, lane);
    a0 = wmma_bf16(a, b0, a0);
    v16bf b1 = make_bfrag([&](int k, int n) {
      const int kk = k0 + k;
      return (kk < kAD) ? lkB[kk * kAD + n0 + n] : (__bf16)0.f; }, lane);
    a1 = wmma_bf16(a, b1, a1);
    v16bf b2 = make_bfrag([&](int k, int n) {
      const int kk = k0 + k;
      return (kk < kAD) ? lvB[kk * kAD + n0 + n] : (__bf16)0.f; }, lane);
    a2 = wmma_bf16(a, b2, a2);
  }
  store_cfrag(a0, lane, [&](int m, int n, float v) {
    lq[(m0 + m) * kAD + n0 + n] = (__bf16)elu(v + lqb[n0 + n]); });
  store_cfrag(a1, lane, [&](int m, int n, float v) {
    lk[(m0 + m) * kAD + n0 + n] = (__bf16)elu(v + lkb[n0 + n]); });
  store_cfrag(a2, lane, [&](int m, int n, float v) {
    lv[(m0 + m) * kAD + n0 + n] = (__bf16)elu(v + lvb[n0 + n]); });
}

// =====================================================================
// Kernel 3: line attention, one wave per (doc, head); 128x128 scores in LDS
// grid = 128 blocks x 32 threads, dynamic LDS = 98304 B
// =====================================================================
__global__ __launch_bounds__(32) void lineattn_kernel(
    const __bf16* __restrict__ lq, const __bf16* __restrict__ lk,
    const __bf16* __restrict__ lv, const int* __restrict__ maskL,
    __bf16* __restrict__ latt) {
  extern __shared__ char smemL[];
  float*  sS = (float*)smemL;
  __bf16* sW = (__bf16*)(sS + kLINES * kLINES);
  const int doc = blockIdx.x >> 3, h = blockIdx.x & 7;
  const int lane = threadIdx.x;
  const int base = doc * kLINES;

  for (int mt = 0; mt < 8; ++mt)
    for (int nt = 0; nt < 8; ++nt) {
      const int m0 = mt * 16, n0 = nt * 16;
      v8f acc = {};
      for (int kc = 0; kc < 2; ++kc) {          // D = 50 padded to 64
        const int k0 = kc * 32;
        v16bf a = make_afrag([&](int m, int k) {
          const int kk = k0 + k;
          return (kk < kDPH) ? lq[(base + m0 + m) * kAD + h * kDPH + kk] : (__bf16)0.f; }, lane);
        v16bf b = make_bfrag([&](int k, int n) {
          const int kk = k0 + k;
          return (kk < kDPH) ? lk[(base + n0 + n) * kAD + h * kDPH + kk] : (__bf16)0.f; }, lane);
        acc = wmma_bf16(a, b, acc);
      }
      store_cfrag(acc, lane, [&](int m, int n, float v) {
        float s = v * kINVS;
        if (!maskL[base + n0 + n]) s -= kNEG;
        sS[(m0 + m) * kLINES + n0 + n] = s;
      });
    }
  __syncthreads();
  for (int r = lane; r < kLINES; r += 32) {
    float mx = -3.4e38f;
    for (int j = 0; j < kLINES; ++j) mx = fmaxf(mx, sS[r * kLINES + j]);
    float sum = 0.f;
    for (int j = 0; j < kLINES; ++j) sum += __expf(sS[r * kLINES + j] - mx);
    const float rs = (maskL[base + r] ? 1.f : 0.f) / sum;
    for (int j = 0; j < kLINES; ++j)
      sW[r * kLINES + j] = (__bf16)(__expf(sS[r * kLINES + j] - mx) * rs);
  }
  __syncthreads();
  for (int mt = 0; mt < 8; ++mt)
    for (int nt = 0; nt < 4; ++nt) {            // N = 50 padded to 64
      const int m0 = mt * 16, n0 = nt * 16;
      v8f acc = {};
      for (int kc = 0; kc < 4; ++kc) {          // K = 128
        const int k0 = kc * 32;
        v16bf a = make_afrag([&](int m, int k) {
          return sW[(m0 + m) * kLINES + k0 + k]; }, lane);
        v16bf b = make_bfrag([&](int k, int n) {
          const int nn = n0 + n;
          return (nn < kDPH) ? lv[(base + k0 + k) * kAD + h * kDPH + nn] : (__bf16)0.f; }, lane);
        acc = wmma_bf16(a, b, acc);
      }
      store_cfrag(acc, lane, [&](int m, int n, float v) {
        const int nn = n0 + n;
        if (nn < kDPH) {
          const float qm = maskL[base + m0 + m] ? 1.f : 0.f;
          latt[((doc * kHEADS + h) * kLINES + m0 + m) * kDPH + nn] = (__bf16)(v * qm);
        }
      });
    }
}

// =====================================================================
// Kernel 4: doc target attention (wave==head) + fc1/fc2 heads
// =====================================================================
__global__ __launch_bounds__(256) void doc_kernel(
    const __bf16* __restrict__ lk, const __bf16* __restrict__ latt,
    const int* __restrict__ maskL, const float* __restrict__ ltgt,
    const float* __restrict__ fc1w, const float* __restrict__ fc1b,
    const float* __restrict__ fc2w, const float* __restrict__ fc2b,
    float* __restrict__ out) {
  __shared__ float swt[kHEADS * kLINES];
  __shared__ float sdoc[kAD];
  const int doc = blockIdx.x;
  const int tid = threadIdx.x, lane = tid & 31, h = tid >> 5;

  float sv[4];
  float mx = -3.4e38f;
  #pragma unroll
  for (int i = 0; i < 4; ++i) {
    const int j = lane + 32 * i;
    float s = 0.f;
    for (int d = 0; d < kDPH; ++d)
      s += ltgt[h * kDPH + d] * (float)lk[(doc * kLINES + j) * kAD + h * kDPH + d];
    s *= kINVS;
    if (!maskL[doc * kLINES + j]) s -= kNEG;
    sv[i] = s;
    mx = fmaxf(mx, s);
  }
  #pragma unroll
  for (int off = 16; off > 0; off >>= 1) mx = fmaxf(mx, __shfl_xor(mx, off, 32));
  float sum = 0.f;
  #pragma unroll
  for (int i = 0; i < 4; ++i) { sv[i] = __expf(sv[i] - mx); sum += sv[i]; }
  #pragma unroll
  for (int off = 16; off > 0; off >>= 1) sum += __shfl_xor(sum, off, 32);
  const float rs = 1.f / sum;
  #pragma unroll
  for (int i = 0; i < 4; ++i) swt[h * kLINES + lane + 32 * i] = sv[i] * rs;
  __syncthreads();

  for (int idx = tid; idx < kAD; idx += 256) {
    const int hh = idx / kDPH, d = idx - hh * kDPH;
    const __bf16* lp = latt + (doc * kHEADS + hh) * kLINES * kDPH + d;
    float acc = 0.f;
    for (int j = 0; j < kLINES; ++j) acc += swt[hh * kLINES + j] * (float)lp[j * kDPH];
    sdoc[idx] = acc;
  }
  __syncthreads();

  for (int o = tid; o < 582; o += 256) {
    float acc;
    if (o < 70) {
      acc = fc1b[o];
      for (int d = 0; d < kAD; ++d) acc += sdoc[d] * fc1w[d * 70 + o];
    } else {
      const int o2 = o - 70;
      acc = fc2b[o2];
      for (int d = 0; d < kAD; ++d) acc += sdoc[d] * fc2w[d * 512 + o2];
    }
    out[doc * 582 + o] = acc;
  }
}

// =====================================================================
extern "C" void kernel_launch(void* const* d_in, const int* in_sizes, int n_in,
                              void* d_out, int out_size, void* d_ws, size_t ws_size,
                              hipStream_t stream) {
  const int*   docs = (const int*)d_in[0];
  const float* emb  = (const float*)d_in[1];
  const float* wqw = (const float*)d_in[2],  *wqb = (const float*)d_in[3];
  const float* wkw = (const float*)d_in[4],  *wkb = (const float*)d_in[5];
  const float* wvw = (const float*)d_in[6],  *wvb = (const float*)d_in[7];
  const float* wtgt = (const float*)d_in[8];
  const float* lqw = (const float*)d_in[9],  *lqb = (const float*)d_in[10];
  const float* lkw = (const float*)d_in[11], *lkb = (const float*)d_in[12];
  const float* lvw = (const float*)d_in[13], *lvb = (const float*)d_in[14];
  const float* ltgt = (const float*)d_in[15];
  const float* fc1w = (const float*)d_in[16], *fc1b = (const float*)d_in[17];
  const float* fc2w = (const float*)d_in[18], *fc2b = (const float*)d_in[19];

  char* ws = (char*)d_ws;
  __bf16* lineEb = (__bf16*)(ws + 0);          // 2048*400*2 = 819,200
  int*    maskL  = (int*)(ws + 819200);        // 8,192
  __bf16* lq     = (__bf16*)(ws + 827392);     // 1,638,400
  __bf16* lk     = (__bf16*)(ws + 2465792);
  __bf16* lv     = (__bf16*)(ws + 4104192);
  __bf16* latt   = (__bf16*)(ws + 5742592);    // 16*8*128*50*2 = 1,638,400
  __bf16* wqB    = (__bf16*)(ws + 7053312);    // 300*400*2 = 240,000
  __bf16* wkB    = (__bf16*)(ws + 7293312);
  __bf16* wvB    = (__bf16*)(ws + 7533312);
  __bf16* lqB    = (__bf16*)(ws + 7773312);    // 400*400*2 = 320,000
  __bf16* lkB    = (__bf16*)(ws + 8093312);
  __bf16* lvB    = (__bf16*)(ws + 8413312);    // end 8,733,312

  const size_t smemW = 131328;
  const size_t smemL = 98304;

  cvt_weights_kernel<<<256, 256, 0, stream>>>(wqw, wkw, wvw, lqw, lkw, lvw,
                                              wqB, wkB, wvB, lqB, lkB, lvB);
  word_kernel<<<kNLINE, 256, smemW, stream>>>(docs, emb, wqB, wqb, wkB, wkb,
                                              wvB, wvb, wtgt, lineEb, maskL);
  lineproj_kernel<<<400, 256, 0, stream>>>(lineEb, lqB, lqb, lkB, lkb, lvB, lvb,
                                           lq, lk, lv);
  lineattn_kernel<<<kBATCH * kHEADS, 32, smemL, stream>>>(lq, lk, lv, maskL, latt);
  doc_kernel<<<kBATCH, 256, 0, stream>>>(lk, latt, maskL, ltgt,
                                         fc1w, fc1b, fc2w, fc2b, (float*)d_out);
  (void)in_sizes; (void)n_in; (void)out_size; (void)ws_size;
}
